// GPRGNN_62723702391572
// MI455X (gfx1250) — compile-verified, matrix operand order checked by reading, not run on hardware
//
#include <hip/hip_runtime.h>
#include <hip/hip_bf16.h>

// ---------------- problem constants (from reference) ----------------
#define NN       100000     // N_NODES
#define NE       1600000    // N_EDGES
#define IN_DIM   512
#define HID      256
#define OUT_DIM  40
#define KHOPS    10

typedef __attribute__((ext_vector_type(16))) _Float16 v16h;
typedef __attribute__((ext_vector_type(8)))  float    v8f;

#define LDS_STRIDE 264   // padded f16 stride (132 dwords -> 4-bank shift/row)

// ---- 16x32 f16 A/B fragment (WMMA layout) converted from an f32 row ----
// lane m (0..15) holds row/col m; half = lane>>4 selects K sub-block:
//   elems 0..7  <-> K = k0 + half*8 + j
//   elems 8..15 <-> K = k0 + 16 + half*8 + j
__device__ __forceinline__ v16h frag_from_f32(const float* __restrict__ rowptr,
                                              int k0, int half) {
  v16h f;
  const int kA = k0 + half * 8;
  const int kB = k0 + 16 + half * 8;
#pragma unroll
  for (int j = 0; j < 8; ++j) {
    f[j]     = (_Float16)rowptr[kA + j];
    f[j + 8] = (_Float16)rowptr[kB + j];
  }
  return f;
}

// ---- same fragment from an already-f16 row (global f16 weights or LDS) ----
__device__ __forceinline__ v16h frag_from_f16(const _Float16* __restrict__ rowptr,
                                              int k0, int half) {
  v16h f;
  const int kA = k0 + half * 8;
  const int kB = k0 + 16 + half * 8;
#pragma unroll
  for (int j = 0; j < 8; ++j) {
    f[j]     = rowptr[kA + j];
    f[j + 8] = rowptr[kB + j];
  }
  return f;
}

// ============ one-shot weight down-convert: W1,W2 fp32 -> f16 (L2 resident) ======
__global__ void cvt_weights_kernel(const float* __restrict__ W1,
                                   const float* __restrict__ W2,
                                   _Float16* __restrict__ w1h,
                                   _Float16* __restrict__ w2h) {
  int i = blockIdx.x * blockDim.x + threadIdx.x;
  if (i < HID * IN_DIM)  w1h[i] = (_Float16)W1[i];
  if (i < OUT_DIM * HID) w2h[i] = (_Float16)W2[i];
}

// ================= fused MLP encoder: H2 = relu(X W1^T + b1) W2^T + b2 ==========
// One block = one 32-row slab (NN % 32 == 0).
// Stage 1: 8 waves, each 2 M-tiles x 2 N-tiles (4 indep WMMA chains) ->
//          32x256 hidden tile in LDS (f16, already in A-fragment-friendly layout).
// Stage 2: 6 waves, one 16x16 tile each -> 32x48 output (cols >= 40 masked).
__global__ __launch_bounds__(256)
void gprgnn_mlp_wmma(const float* __restrict__ X,
                     const _Float16* __restrict__ w1h,
                     const float* __restrict__ bias1,
                     const _Float16* __restrict__ w2h,
                     const float* __restrict__ bias2,
                     float* __restrict__ H2) {
  __shared__ _Float16 hlds[32 * LDS_STRIDE];

  const int lane    = threadIdx.x & 31;
  const int wave    = threadIdx.x >> 5;
  const int m       = lane & 15;
  const int half    = lane >> 4;
  const int rowBase = blockIdx.x * 32;

  // ---------------- stage 1: hidden = relu(X @ W1^T + b1) ----------------
  {
    const int colBase = wave * 32;            // 8 waves cover 256 hidden cols
    const float*    arow0 = X   + (size_t)(rowBase + m) * IN_DIM;
    const float*    arow1 = X   + (size_t)(rowBase + 16 + m) * IN_DIM;
    const _Float16* brow0 = w1h + (size_t)(colBase + m) * IN_DIM;    // n = lane&15
    const _Float16* brow1 = w1h + (size_t)(colBase + 16 + m) * IN_DIM;
    v8f acc00 = {}, acc01 = {}, acc10 = {}, acc11 = {};
#pragma unroll 2
    for (int k0 = 0; k0 < IN_DIM; k0 += 32) {
      __builtin_prefetch(arow0 + k0 + 64, 0, 0);  // global_prefetch_b8
      v16h a0 = frag_from_f32(arow0, k0, half);
      v16h a1 = frag_from_f32(arow1, k0, half);
      v16h b0 = frag_from_f16(brow0, k0, half);
      v16h b1 = frag_from_f16(brow1, k0, half);
      // 4 independent accumulator chains: A and B fragments each reused twice
      acc00 = __builtin_amdgcn_wmma_f32_16x16x32_f16(false, a0, false, b0,
                                                     (short)0, acc00, false, false);
      acc01 = __builtin_amdgcn_wmma_f32_16x16x32_f16(false, a0, false, b1,
                                                     (short)0, acc01, false, false);
      acc10 = __builtin_amdgcn_wmma_f32_16x16x32_f16(false, a1, false, b0,
                                                     (short)0, acc10, false, false);
      acc11 = __builtin_amdgcn_wmma_f32_16x16x32_f16(false, a1, false, b1,
                                                     (short)0, acc11, false, false);
    }
    const int c0 = colBase + m;
    const int c1 = c0 + 16;
    const float bb0 = bias1[c0];
    const float bb1 = bias1[c1];
#pragma unroll
    for (int r = 0; r < 8; ++r) {
      const int row = r + 8 * half;           // C/D layout: VGPR r <-> M = r + 8*half
      float v00 = acc00[r] + bb0; v00 = v00 > 0.f ? v00 : 0.f;
      float v01 = acc01[r] + bb1; v01 = v01 > 0.f ? v01 : 0.f;
      float v10 = acc10[r] + bb0; v10 = v10 > 0.f ? v10 : 0.f;
      float v11 = acc11[r] + bb1; v11 = v11 > 0.f ? v11 : 0.f;
      hlds[row * LDS_STRIDE + c0]        = (_Float16)v00;
      hlds[row * LDS_STRIDE + c1]        = (_Float16)v01;
      hlds[(row + 16) * LDS_STRIDE + c0] = (_Float16)v10;
      hlds[(row + 16) * LDS_STRIDE + c1] = (_Float16)v11;
    }
  }
  __syncthreads();

  // ---------------- stage 2: H2 = hidden @ W2^T + b2 ----------------
  if (wave < 6) {                             // 2 M-tiles x 3 N-tiles (48 >= 40)
    const int mt = wave / 3;
    const int nt = wave % 3;
    const int n     = nt * 16 + m;
    const int nclip = n < OUT_DIM ? n : (OUT_DIM - 1);     // keep loads in-bounds
    const _Float16* brow = w2h  + (size_t)nclip * HID;
    const _Float16* arow = hlds + (mt * 16 + m) * LDS_STRIDE;
    v8f acc = {};
#pragma unroll
    for (int k0 = 0; k0 < HID; k0 += 32) {
      v16h a = frag_from_f16(arow, k0, half);
      v16h b = frag_from_f16(brow, k0, half);
      acc = __builtin_amdgcn_wmma_f32_16x16x32_f16(false, a, false, b,
                                                   (short)0, acc, false, false);
    }
    if (n < OUT_DIM) {
      const float bb = bias2[n];
#pragma unroll
      for (int r = 0; r < 8; ++r) {
        const int row = rowBase + mt * 16 + r + 8 * half;
        H2[(size_t)row * OUT_DIM + n] = acc[r] + bb;
      }
    }
  }
}

// ================= degree / normalization ==================
__global__ void deg_init_kernel(float* __restrict__ deg) {
  int i = blockIdx.x * blockDim.x + threadIdx.x;
  if (i < NN) deg[i] = 1.0f;                  // self loop
}

__global__ void deg_edge_kernel(const int* __restrict__ dst,
                                float* __restrict__ deg) {
  int e = blockIdx.x * blockDim.x + threadIdx.x;
  if (e < NE) atomicAdd(&deg[dst[e]], 1.0f);
}

__global__ void dinv_kernel(float* __restrict__ deg) {
  int i = blockIdx.x * blockDim.x + threadIdx.x;
  if (i < NN) deg[i] = rsqrtf(deg[i]);        // deg >= 1 always
}

// ================= propagation ==================
// z (+)= temp[k]*hin ; hnext = dinv^2 * hin   (self-loop term of the next hop)
__global__ void gpr_axpy_selfinit(const float* __restrict__ hin,
                                  const float* __restrict__ dinv,
                                  const float* __restrict__ temp, int k, int first,
                                  float* __restrict__ z,
                                  float* __restrict__ hnext) {
  int idx = blockIdx.x * blockDim.x + threadIdx.x;
  if (idx >= NN * OUT_DIM) return;
  const int node = idx / OUT_DIM;
  const float v  = hin[idx];
  const float tk = temp[k];
  float zz = tk * v;
  if (!first) zz += z[idx];
  z[idx] = zz;
  const float di = dinv[node];
  hnext[idx] = di * di * v;
}

// hout[dst] += dinv[src]*dinv[dst] * hin[src]  over all edges.
// 8 lanes per edge, each lane handles 5 of the 40 features (stride 8, coalesced).
__global__ void gpr_scatter_edges(const int* __restrict__ src,
                                  const int* __restrict__ dst,
                                  const float* __restrict__ dinv,
                                  const float* __restrict__ hin,
                                  float* __restrict__ hout) {
  int idx = blockIdx.x * blockDim.x + threadIdx.x;
  int e   = idx >> 3;
  int sub = idx & 7;
  if (e >= NE) return;
  const int s = src[e];
  const int d = dst[e];
  const float w = dinv[s] * dinv[d];
  const float* hs = hin  + (size_t)s * OUT_DIM;
  float*       hd = hout + (size_t)d * OUT_DIM;
#pragma unroll
  for (int j = 0; j < 5; ++j) {               // 8*5 == 40 exactly
    const int c = sub + 8 * j;
    atomicAdd(&hd[c], w * hs[c]);
  }
}

// ================= host-side orchestration ==================
extern "C" void kernel_launch(void* const* d_in, const int* in_sizes, int n_in,
                              void* d_out, int out_size, void* d_ws, size_t ws_size,
                              hipStream_t stream) {
  const float* x     = (const float*)d_in[0];
  const int*   ei    = (const int*)d_in[1];      // [2, NE] int32
  const float* W1    = (const float*)d_in[2];
  const float* bias1 = (const float*)d_in[3];
  const float* W2    = (const float*)d_in[4];
  const float* bias2 = (const float*)d_in[5];
  const float* temp  = (const float*)d_in[6];
  float*       z     = (float*)d_out;            // [NN, OUT_DIM]

  const int* src = ei;
  const int* dst = ei + NE;

  // workspace: dinv[NN] | bufA[NN*40] | bufB[NN*40] | w1h | w2h  (~32.7 MB)
  float* dinv = (float*)d_ws;
  float* bufA = dinv + NN;
  float* bufB = bufA + (size_t)NN * OUT_DIM;
  _Float16* w1h = (_Float16*)(bufB + (size_t)NN * OUT_DIM);
  _Float16* w2h = w1h + (size_t)HID * IN_DIM;

  // 0) one-shot weight down-convert (f16 weights stay L2 resident)
  cvt_weights_kernel<<<(HID * IN_DIM + 255) / 256, 256, 0, stream>>>(W1, W2, w1h, w2h);

  // 1) fused WMMA MLP encoder -> bufA = h  (hidden never touches HBM)
  gprgnn_mlp_wmma<<<NN / 32, 256, 0, stream>>>(x, w1h, bias1, w2h, bias2, bufA);

  // 2) gcn_norm: dinv = rsqrt(in-degree + 1)
  deg_init_kernel<<<(NN + 255) / 256, 256, 0, stream>>>(dinv);
  deg_edge_kernel<<<(NE + 255) / 256, 256, 0, stream>>>(dst, dinv);
  dinv_kernel<<<(NN + 255) / 256, 256, 0, stream>>>(dinv);

  // 3) GPR propagation: z = sum_k temp[k] * A_hat^k h
  const int nElem   = NN * OUT_DIM;
  const int axpyGrd = (nElem + 255) / 256;
  const int scatGrd = (NE * 8 + 255) / 256;

  // k = 0: z = temp[0]*h ; bufB = self-loop init of hop 1
  gpr_axpy_selfinit<<<axpyGrd, 256, 0, stream>>>(bufA, dinv, temp, 0, 1, z, bufB);

  float* cur = bufA;   // holds h_{k-1}
  float* nxt = bufB;   // pre-initialized with dinv^2 * h_{k-1}
  for (int k = 1; k <= KHOPS; ++k) {
    gpr_scatter_edges<<<scatGrd, 256, 0, stream>>>(src, dst, dinv, cur, nxt);
    // z += temp[k] * h_k ; cur <- self-loop init for hop k+1
    gpr_axpy_selfinit<<<axpyGrd, 256, 0, stream>>>(nxt, dinv, temp, k, 0, z, cur);
    float* t = cur; cur = nxt; nxt = t;
  }
}
